// Head_1168231105030
// MI455X (gfx1250) — matrix operand order repaired and checked by reading, not compiled, not run on hardware
//
#include <hip/hip_runtime.h>

typedef __attribute__((ext_vector_type(16))) _Float16 v16h;
typedef __attribute__((ext_vector_type(8)))  _Float16 v8h;
typedef __attribute__((ext_vector_type(8)))  float    v8f;
typedef __attribute__((ext_vector_type(4)))  float    v4f;

#define B_BATCH 4
#define T_SEQ   4096
#define E_DIM   512
#define H_DIM   64

static __device__ __forceinline__ v8f wmma_f16(v16h a, v16h b, v8f c) {
  // (neg_a, A, neg_b, B, c_mod, C, reuse_a, reuse_b)
  return __builtin_amdgcn_wmma_f32_16x16x32_f16(false, a, false, b, (short)0, c,
                                                false, false);
}

static __device__ __forceinline__ v16h cat16(v8h lo, v8h hi) {
  v16h r;
#pragma unroll
  for (int i = 0; i < 8; ++i) { r[i] = lo[i]; r[i + 8] = hi[i]; }
  return r;
}

// Async global->LDS copy, 16 bytes per lane (CDNA5, tracked by ASYNCcnt).
// VDST = per-lane LDS byte address (generic LDS pointers carry the LDS offset
// in their low 32 bits per the aperture mapping), VADDR = 64-bit global addr.
static __device__ __forceinline__ void async_copy_b128(const _Float16* gsrc,
                                                       _Float16* ldst) {
  const unsigned loff = (unsigned)(size_t)ldst;
  asm volatile("global_load_async_to_lds_b128 %0, %1, off"
               :: "v"(loff), "v"(gsrc)
               : "memory");
}

static __device__ __forceinline__ void wait_async_le1() {
  asm volatile("s_wait_asynccnt 0x1" ::: "memory");
}
static __device__ __forceinline__ void wait_async_0() {
  asm volatile("s_wait_asynccnt 0x0" ::: "memory");
}

// ---------------------------------------------------------------------------
// Projection: Y[BT x 64] (f16) = X[BT x 512] (f32) @ W[512 x 64] (f32)
// Block = 256 threads = 8 waves; each wave -> 16 rows x 64 cols.
// W^T staged in LDS in two 256-deep phases (32 KB + pad).
// ---------------------------------------------------------------------------
__global__ __launch_bounds__(256) void proj_f16_kernel(
    const float* __restrict__ X, const float* __restrict__ W,
    _Float16* __restrict__ Y) {
  __shared__ _Float16 wt[H_DIM][264];  // W^T chunk [h][e_local], 256 used + pad

  const int tid    = threadIdx.x;
  const int lane   = tid & 31;
  const int w      = tid >> 5;
  const int half16 = lane >> 4;
  const int ln     = lane & 15;
  const int rowBlk = blockIdx.x * 128;
  const int mRow   = rowBlk + w * 16 + ln;  // A row this lane feeds

  v8f acc[4] = {};

  for (int ph = 0; ph < 2; ++ph) {
    __syncthreads();
    {  // cooperative W^T load: thread tid owns W row (ph*256 + tid)
      const float* wrow = W + (size_t)(ph * 256 + tid) * H_DIM;
#pragma unroll
      for (int h = 0; h < H_DIM; h += 4) {
        v4f wv = *(const v4f*)(wrow + h);
#pragma unroll
        for (int i = 0; i < 4; ++i) wt[h + i][tid] = (_Float16)wv[i];
      }
    }
    __syncthreads();

#pragma unroll
    for (int kit = 0; kit < 8; ++kit) {
      const int k0 = ph * 256 + kit * 32;
      // A fragment: a[0..7]=X[m][kb..kb+7], a[8..15]=X[m][kb+16..kb+23]
      const float* ar = X + (size_t)mRow * E_DIM + k0 + half16 * 8;
      v4f a0 = *(const v4f*)(ar);
      v4f a1 = *(const v4f*)(ar + 4);
      v4f a2 = *(const v4f*)(ar + 16);
      v4f a3 = *(const v4f*)(ar + 20);
      v16h af;
#pragma unroll
      for (int i = 0; i < 4; ++i) {
        af[i]      = (_Float16)a0[i];
        af[4 + i]  = (_Float16)a1[i];
        af[8 + i]  = (_Float16)a2[i];
        af[12 + i] = (_Float16)a3[i];
      }
      const int eb = kit * 32 + half16 * 8;
#pragma unroll
      for (int t = 0; t < 4; ++t) {
        const _Float16* bp = &wt[t * 16 + ln][eb];
        v16h bf = cat16(*(const v8h*)bp, *(const v8h*)(bp + 16));
        acc[t] = wmma_f16(af, bf, acc[t]);
      }
    }
  }

#pragma unroll
  for (int t = 0; t < 4; ++t)
#pragma unroll
    for (int r = 0; r < 8; ++r) {
      const int row = rowBlk + w * 16 + r + 8 * half16;
      Y[(size_t)row * H_DIM + t * 16 + ln] = (_Float16)acc[t][r];
    }
}

// ---------------------------------------------------------------------------
// Flash attention (causal). Block = 128 query rows (8 waves x 16 rows),
// iterating 32-key blocks. K tiles: double-buffered ASYNC global->LDS copies
// (no VGPR staging). V tiles: software-pipelined VGPR load + transposed LDS
// store (transpose needed so PV B-fragments are contiguous b128 LDS reads).
// ---------------------------------------------------------------------------
__global__ __launch_bounds__(256) void attn_kernel(
    const _Float16* __restrict__ Qp, const _Float16* __restrict__ Kp,
    const _Float16* __restrict__ Vp, float* __restrict__ Out) {
  __shared__ _Float16 kls[2][32][72];   // key block, row-major [key][h]
  __shared__ _Float16 vtls[2][64][40];  // value block transposed [h][key]
  __shared__ _Float16 pls[8][16][40];   // per-wave P tile [m][key]

  const int tid    = threadIdx.x;
  const int lane   = tid & 31;
  const int w      = tid >> 5;
  const int half16 = lane >> 4;
  const int ln     = lane & 15;
  const int qbase  = blockIdx.x * 128;       // query tile base within batch
  const int boff   = blockIdx.y * T_SEQ;     // batch row offset
  const float scale = 0.044194173824159216f; // 1/sqrt(E=512)

  // Q fragments (persist whole kernel): rows w*16+ln of this tile, K-dim = H
  const _Float16* qptr = Qp + (size_t)(boff + qbase + w * 16 + ln) * H_DIM;
  v16h qfrag[2];
#pragma unroll
  for (int kk = 0; kk < 2; ++kk) {
    const int base = kk * 32 + half16 * 16;
    qfrag[kk] = cat16(*(const v8h*)(qptr + base), *(const v8h*)(qptr + base + 8));
  }

  v8f o[4] = {};
  float mstate[8], lstate[8];
#pragma unroll
  for (int r = 0; r < 8; ++r) { mstate[r] = -3.0e38f; lstate[r] = 0.0f; }

  const int numKB = qbase / 32 + 4;  // causal: keys 0 .. qbase+127

  // Per-thread slice of each 32x64 tile: one 16-byte chunk.
  const int kr = tid >> 3;        // key row 0..31
  const int hc = (tid & 7) * 8;   // h chunk 0..56

  // Prologue: async K tile 0 -> buffer 0; V tile 0 -> VGPRs.
  async_copy_b128(Kp + (size_t)(boff + kr) * H_DIM + hc, &kls[0][kr][hc]);
  v8h vv = *(const v8h*)(Vp + (size_t)(boff + kr) * H_DIM + hc);

  for (int jb = 0; jb < numKB; ++jb) {
    const int j0  = jb * 32;
    const int cur = jb & 1;

    // Issue async K load for the NEXT tile into the other buffer (its previous
    // readers all passed the barrier that ended iteration jb-1).
    if (jb + 1 < numKB) {
      const size_t gn = (size_t)(boff + j0 + 32 + kr) * H_DIM + hc;
      async_copy_b128(Kp + gn, &kls[cur ^ 1][kr][hc]);
    }

    // Store this tile's V (loaded last iteration) transposed into LDS.
#pragma unroll
    for (int i = 0; i < 8; ++i) vtls[cur][hc + i][kr] = vv[i];

    if (jb + 1 < numKB) {
      // Prefetch next V tile into VGPRs, then retire this tile's K async copy
      // (leave exactly the next tile's 1 async op per wave in flight).
      vv = *(const v8h*)(Vp + (size_t)(boff + j0 + 32 + kr) * H_DIM + hc);
      wait_async_le1();
    } else {
      wait_async_0();
    }
    __syncthreads();

    // S(16x32) = Q(16x64) @ K^T : two N-tiles, K-loop over H in 2 steps
    v8f s0 = {}, s1 = {};
#pragma unroll
    for (int kk = 0; kk < 2; ++kk) {
      const int kb = kk * 32 + half16 * 16;
      const _Float16* b0 = &kls[cur][ln][kb];
      const _Float16* b1 = &kls[cur][16 + ln][kb];
      v16h bf0 = cat16(*(const v8h*)b0, *(const v8h*)(b0 + 8));
      v16h bf1 = cat16(*(const v8h*)b1, *(const v8h*)(b1 + 8));
      s0 = wmma_f16(qfrag[kk], bf0, s0);
      s1 = wmma_f16(qfrag[kk], bf1, s1);
    }

    // online softmax; C layout: VGPR r -> row r + 8*half16, col = ln (+16)
    const int col0 = j0 + ln;
    float facs[8];
#pragma unroll
    for (int r = 0; r < 8; ++r) {
      const int qi = qbase + w * 16 + r + 8 * half16;
      float v0 = s0[r] * scale;
      float v1 = s1[r] * scale;
      if (col0 > qi) v0 = -3.0e38f;
      if (col0 + 16 > qi) v1 = -3.0e38f;
      float mx = fmaxf(v0, v1);
#pragma unroll
      for (int off = 8; off >= 1; off >>= 1)
        mx = fmaxf(mx, __shfl_xor(mx, off, 32));
      const float mnew = fmaxf(mstate[r], mx);
      const float p0 = __expf(v0 - mnew);
      const float p1 = __expf(v1 - mnew);
      float rs = p0 + p1;
#pragma unroll
      for (int off = 8; off >= 1; off >>= 1) rs += __shfl_xor(rs, off, 32);
      const float fac = __expf(mstate[r] - mnew);
      lstate[r] = lstate[r] * fac + rs;
      mstate[r] = mnew;
      facs[r] = fac;
      const int prow = r + 8 * half16;
      pls[w][prow][ln]      = (_Float16)p0;   // same-wave DS ops are in-order
      pls[w][prow][ln + 16] = (_Float16)p1;
    }

#pragma unroll
    for (int t = 0; t < 4; ++t)
#pragma unroll
      for (int r = 0; r < 8; ++r) o[t][r] *= facs[r];

    // O += P(16x32) @ V(32x64): A-layout fragment re-read from LDS
    const _Float16* pp = &pls[w][ln][half16 * 16];
    v16h pfrag = cat16(*(const v8h*)pp, *(const v8h*)(pp + 8));
    const int kb = half16 * 16;
#pragma unroll
    for (int t = 0; t < 4; ++t) {
      const _Float16* vp2 = &vtls[cur][t * 16 + ln][kb];
      v16h vf = cat16(*(const v8h*)vp2, *(const v8h*)(vp2 + 8));
      o[t] = wmma_f16(pfrag, vf, o[t]);
    }
    __syncthreads();  // all waves done reading buffers before next overwrite
  }

  // epilogue: normalize and store f32
  float inv[8];
#pragma unroll
  for (int r = 0; r < 8; ++r) inv[r] = 1.0f / lstate[r];
#pragma unroll
  for (int t = 0; t < 4; ++t)
#pragma unroll
    for (int r = 0; r < 8; ++r) {
      const int row = boff + qbase + w * 16 + r + 8 * half16;
      Out[(size_t)row * H_DIM + t * 16 + ln] = o[t][r] * inv[r];
    }
}

// ---------------------------------------------------------------------------
extern "C" void kernel_launch(void* const* d_in, const int* in_sizes, int n_in,
                              void* d_out, int out_size, void* d_ws,
                              size_t ws_size, hipStream_t stream) {
  (void)in_sizes; (void)n_in; (void)out_size; (void)ws_size;
  const float* v  = (const float*)d_in[0];
  const float* k  = (const float*)d_in[1];
  const float* q  = (const float*)d_in[2];
  const float* Wv = (const float*)d_in[3];
  const float* Wk = (const float*)d_in[4];
  const float* Wq = (const float*)d_in[5];
  // d_in[6] = causal mask: replaced by analytic (col > row) test.
  float* out = (float*)d_out;

  const size_t n = (size_t)B_BATCH * T_SEQ * H_DIM;  // 1,048,576 elems
  _Float16* Qp = (_Float16*)d_ws;
  _Float16* Kp = Qp + n;
  _Float16* Vp = Qp + 2 * n;

  const dim3 pgrid((B_BATCH * T_SEQ) / 128);
  proj_f16_kernel<<<pgrid, 256, 0, stream>>>(q, Wq, Qp);
  proj_f16_kernel<<<pgrid, 256, 0, stream>>>(k, Wk, Kp);
  proj_f16_kernel<<<pgrid, 256, 0, stream>>>(v, Wv, Vp);

  attn_kernel<<<dim3(T_SEQ / 128, B_BATCH), 256, 0, stream>>>(Qp, Kp, Vp, out);
}
